// GPT2SelfAttention_32950989095337
// MI455X (gfx1250) — matrix operand order repaired and checked by reading, not compiled, not run on hardware
//
#include <hip/hip_runtime.h>
#include <stdint.h>

// ---------- types ----------
typedef __bf16 bf16_t;
typedef bf16_t v16bf __attribute__((ext_vector_type(16)));
typedef float  v8f   __attribute__((ext_vector_type(8)));

struct alignas(16) U4 { unsigned int x, y, z, w; };
struct alignas(16) F4 { float x, y, z, w; };
struct alignas(8)  U2 { unsigned int x, y; };

// float -> bf16 (round to nearest even)
static __device__ __forceinline__ unsigned short f2bf(float f) {
    unsigned int u = __float_as_uint(f);
    u += 0x7FFFu + ((u >> 16) & 1u);
    return (unsigned short)(u >> 16);
}

// build a v16bf WMMA operand from two 16-byte chunks
static __device__ __forceinline__ v16bf ld_v16(const unsigned short* lo,
                                               const unsigned short* hi) {
    union { U4 q[2]; v16bf v; } u;
    u.q[0] = *(const U4*)lo;
    u.q[1] = *(const U4*)hi;
    return u.v;
}

// constant all-1.0 bf16 operand (for row-sum via WMMA)
static __device__ __forceinline__ v16bf ones_v16() {
    union { unsigned int u[8]; v16bf v; } x;
#pragma unroll
    for (int i = 0; i < 8; ++i) x.u[i] = 0x3F803F80u;
    return x.v;
}

static __device__ __forceinline__ v8f wmma_bf16(v16bf a, v16bf b, v8f c) {
    return __builtin_amdgcn_wmma_f32_16x16x32_bf16(false, a, false, b,
                                                   (short)0, c, false, false);
}

// =====================================================================
// Tiled bf16 GEMM with f32 accumulate:  C[M,N] = A[M,K] * B[K,N] + bias
// 256 threads = 8 waves (2 M x 4 N). Wave tile 32x64, WG tile 64x256.
// Software-pipelined: tile k+1 global loads issued before the compute
// barrier so they fly under tile k's WMMA chain. LDS rows padded to 40
// shorts (80B) -> conflict-free, 16B-aligned b128 operand loads.
// =====================================================================
template <bool OUT_BF16>
__global__ __launch_bounds__(256) void gemm_bias_kernel(
    const float* __restrict__ A, const float* __restrict__ Bw,
    const float* __restrict__ bias, void* __restrict__ Cout,
    int M, int N, int K) {
    __shared__ unsigned short As[64 * 40];   // [m][k]
    __shared__ unsigned short Bs[256 * 40];  // [n][k] (transposed)

    const int tid  = threadIdx.x;
    const int lane = tid & 31;
    const int wid  = tid >> 5;
    const int waveM = wid >> 2;  // 0..1
    const int waveN = wid & 3;   // 0..3
    const int mBase = blockIdx.y * 64;
    const int nBase = blockIdx.x * 256;
    const int l16 = lane & 15;
    const int cA  = (lane < 16) ? 0 : 8;    // A-layout: K {c..c+7, c+16..c+23}
    const int cB  = (lane < 16) ? 0 : 8;    // B-layout: K {c..c+7, c+16..c+23}

    v8f acc[2][4];
#pragma unroll
    for (int s = 0; s < 2; ++s)
#pragma unroll
        for (int t = 0; t < 4; ++t)
#pragma unroll
            for (int j = 0; j < 8; ++j) acc[s][t][j] = 0.0f;

    // staging registers (pipeline)
    F4 aReg[2], bReg[8];
    auto load_tile = [&](int k0) {
#pragma unroll
        for (int i = 0; i < 2; ++i) {
            int idx = tid + 256 * i;
            int r = idx >> 3, c4 = (idx & 7) * 4;
            aReg[i] = *(const F4*)&A[(size_t)(mBase + r) * K + (k0 + c4)];
        }
#pragma unroll
        for (int i = 0; i < 8; ++i) {
            int idx = tid + 256 * i;
            int kk = idx >> 6, n4 = (idx & 63) * 4;
            bReg[i] = *(const F4*)&Bw[(size_t)(k0 + kk) * N + (nBase + n4)];
        }
    };

    load_tile(0);
    for (int k0 = 0; k0 < K; k0 += 32) {
        __syncthreads();
        // store staged tile (f32 -> bf16)
#pragma unroll
        for (int i = 0; i < 2; ++i) {
            int idx = tid + 256 * i;
            int r = idx >> 3, c4 = (idx & 7) * 4;
            U2 p;
            p.x = (unsigned)f2bf(aReg[i].x) | ((unsigned)f2bf(aReg[i].y) << 16);
            p.y = (unsigned)f2bf(aReg[i].z) | ((unsigned)f2bf(aReg[i].w) << 16);
            *(U2*)&As[r * 40 + c4] = p;
        }
#pragma unroll
        for (int i = 0; i < 8; ++i) {
            int idx = tid + 256 * i;
            int kk = idx >> 6, n4 = (idx & 63) * 4;
            Bs[(n4 + 0) * 40 + kk] = f2bf(bReg[i].x);
            Bs[(n4 + 1) * 40 + kk] = f2bf(bReg[i].y);
            Bs[(n4 + 2) * 40 + kk] = f2bf(bReg[i].z);
            Bs[(n4 + 3) * 40 + kk] = f2bf(bReg[i].w);
        }
        // issue next tile's global loads: they overlap the WMMA chain below
        if (k0 + 32 < K) load_tile(k0 + 32);
        __syncthreads();

        v16bf aOp[2], bOp[4];
#pragma unroll
        for (int s = 0; s < 2; ++s) {
            int r = waveM * 32 + s * 16 + l16;
            aOp[s] = ld_v16(&As[r * 40 + cA], &As[r * 40 + cA + 16]);
        }
#pragma unroll
        for (int t = 0; t < 4; ++t) {
            int n = waveN * 64 + t * 16 + l16;
            bOp[t] = ld_v16(&Bs[n * 40 + cB], &Bs[n * 40 + cB + 16]);
        }
#pragma unroll
        for (int s = 0; s < 2; ++s)
#pragma unroll
            for (int t = 0; t < 4; ++t)
                acc[s][t] = wmma_bf16(aOp[s], bOp[t], acc[s][t]);
    }

    // epilogue: + bias, store
    const int rHi = 8 * (lane >> 4);
#pragma unroll
    for (int t = 0; t < 4; ++t) {
        int nG = nBase + waveN * 64 + t * 16 + l16;
        float bv = bias[nG];
#pragma unroll
        for (int s = 0; s < 2; ++s) {
#pragma unroll
            for (int j = 0; j < 8; ++j) {
                int mG = mBase + waveM * 32 + s * 16 + j + rHi;
                float v = acc[s][t][j] + bv;
                if (OUT_BF16)
                    ((unsigned short*)Cout)[(size_t)mG * N + nG] = f2bf(v);
                else
                    ((float*)Cout)[(size_t)mG * N + nG] = v;
            }
        }
    }
}

// =====================================================================
// Flash attention: one WG (4 waves, 128 thr) per (b, h, 64-query block).
// Each wave owns 16 query rows; 64-key blocks (4 score tiles).
// Softmax denominator accumulated via WMMA with an all-ones B operand
// (no sum shuffles); only the row-max uses a 4-step butterfly.
// =====================================================================
__global__ __launch_bounds__(128) void attention_kernel(
    const unsigned short* __restrict__ qkv, float* __restrict__ out) {
    constexpr int S = 2048, D = 1024, HD = 64;
    constexpr int TD = 3 * D;  // 3072
    constexpr int VP = 72;     // vT/pB row pitch (shorts): 144B, 16B-aligned

    __shared__ unsigned short vT[64 * VP];     // [dim][key] (64 keys + pad)
    __shared__ unsigned short pB[4][16 * VP];  // per-wave P tile [row][key]

    const int tid  = threadIdx.x;
    const int lane = tid & 31;
    const int wid  = tid >> 5;
    const int l16  = lane & 15;
    const int cA   = (lane < 16) ? 0 : 8;
    const int cB   = (lane < 16) ? 0 : 8;
    const int rHi  = 8 * (lane >> 4);

    const int b = blockIdx.z, h = blockIdx.y;
    const int qbase = blockIdx.x * 64;
    const int qrow0 = qbase + wid * 16;  // wave's first query row

    // Q as two K=32 A-operands, gathered straight from global bf16
    size_t qoff = ((size_t)(b * S + qrow0 + l16)) * TD + h * HD;
    v16bf qa0 = ld_v16(qkv + qoff + cA,      qkv + qoff + cA + 16);
    v16bf qa1 = ld_v16(qkv + qoff + 32 + cA, qkv + qoff + 32 + cA + 16);
    const v16bf onesB = ones_v16();

    v8f o[4], lacc;
    float m_run[8];
#pragma unroll
    for (int t = 0; t < 4; ++t)
#pragma unroll
        for (int j = 0; j < 8; ++j) o[t][j] = 0.0f;
#pragma unroll
    for (int j = 0; j < 8; ++j) { lacc[j] = 0.0f; m_run[j] = -1e30f; }

    const int kend = qbase + 64;  // causal frontier for this WG
    for (int kb = 0; kb < kend; kb += 64) {
        __syncthreads();
        // stage V^T block: 64 keys x 64 dims -> vT[dim][key]
        // global b128 loads (8 bf16 dims per load), transposed b16 stores
#pragma unroll
        for (int i = 0; i < 4; ++i) {
            int idx = tid + 128 * i;          // 512 U4 chunks total
            int key = idx >> 3, d8 = (idx & 7) * 8;
            U4 u = *(const U4*)&qkv[((size_t)(b * S + kb + key)) * TD + 2 * D +
                                    h * HD + d8];
            vT[(d8 + 0) * VP + key] = (unsigned short)(u.x & 0xffff);
            vT[(d8 + 1) * VP + key] = (unsigned short)(u.x >> 16);
            vT[(d8 + 2) * VP + key] = (unsigned short)(u.y & 0xffff);
            vT[(d8 + 3) * VP + key] = (unsigned short)(u.y >> 16);
            vT[(d8 + 4) * VP + key] = (unsigned short)(u.z & 0xffff);
            vT[(d8 + 5) * VP + key] = (unsigned short)(u.z >> 16);
            vT[(d8 + 6) * VP + key] = (unsigned short)(u.w & 0xffff);
            vT[(d8 + 7) * VP + key] = (unsigned short)(u.w >> 16);
        }
        __syncthreads();

        // prefetch next key block's K and V rows (global_prefetch_b8)
        if (kb + 64 < kend) {
            size_t nrow = ((size_t)(b * S + kb + 64 + (tid & 63))) * TD;
            __builtin_prefetch(&qkv[nrow + D + h * HD], 0, 0);
            __builtin_prefetch(&qkv[nrow + 2 * D + h * HD], 0, 0);
        }

        // scores: four 16x16 tiles over 64 keys
        float st[4][8];
#pragma unroll
        for (int tt = 0; tt < 4; ++tt) {
            size_t koff =
                ((size_t)(b * S + kb + tt * 16 + l16)) * TD + D + h * HD;
            v16bf k0 = ld_v16(qkv + koff + cB,      qkv + koff + cB + 16);
            v16bf k1 = ld_v16(qkv + koff + 32 + cB, qkv + koff + 32 + cB + 16);
            v8f s;
#pragma unroll
            for (int j = 0; j < 8; ++j) s[j] = 0.0f;
            s = wmma_bf16(qa0, k0, s);
            s = wmma_bf16(qa1, k1, s);
#pragma unroll
            for (int j = 0; j < 8; ++j) st[tt][j] = s[j] * 0.125f;  // 1/sqrt(64)
            if (kb + 63 > qrow0) {  // causal mask possible in this tile
                int key = kb + tt * 16 + l16;
#pragma unroll
                for (int j = 0; j < 8; ++j) {
                    int q = qrow0 + j + rHi;
                    if (key > q) st[tt][j] = -10000.0f;
                }
            }
        }

        // online softmax: row-max butterfly (only reduction left)
#pragma unroll
        for (int j = 0; j < 8; ++j) {
            float tmax = fmaxf(fmaxf(st[0][j], st[1][j]),
                               fmaxf(st[2][j], st[3][j]));
#pragma unroll
            for (int msk = 8; msk >= 1; msk >>= 1)
                tmax = fmaxf(tmax, __shfl_xor(tmax, msk, 16));
            float mn = fmaxf(m_run[j], tmax);
            float sc = __expf(m_run[j] - mn);
            m_run[j] = mn;
            lacc[j] *= sc;
#pragma unroll
            for (int t = 0; t < 4; ++t) o[t][j] *= sc;
            int pr = j + rHi;
#pragma unroll
            for (int tt = 0; tt < 4; ++tt) {
                float p = __expf(st[tt][j] - mn);
                pB[wid][pr * VP + tt * 16 + l16] = f2bf(p);
            }
        }
        // wave-private LDS re-layout C->A: ensure stores land before gathers
        asm volatile("s_wait_dscnt 0" ::: "memory");
        v16bf pA0 = ld_v16(&pB[wid][l16 * VP + cA],
                           &pB[wid][l16 * VP + cA + 16]);
        v16bf pA1 = ld_v16(&pB[wid][l16 * VP + 32 + cA],
                           &pB[wid][l16 * VP + 32 + cA + 16]);
#pragma unroll
        for (int t = 0; t < 4; ++t) {
            int d = t * 16 + l16;
            v16bf vOp0 = ld_v16(&vT[d * VP + cB],      &vT[d * VP + cB + 16]);
            v16bf vOp1 = ld_v16(&vT[d * VP + 32 + cB], &vT[d * VP + 32 + cB + 16]);
            o[t] = wmma_bf16(pA0, vOp0, o[t]);
            o[t] = wmma_bf16(pA1, vOp1, o[t]);
        }
        // softmax denominator: P x ones (every lane gets its rows' sums)
        lacc = wmma_bf16(pA0, onesB, lacc);
        lacc = wmma_bf16(pA1, onesB, lacc);
    }

    // epilogue: O / l, write merged-heads f32 [B*S][D]
#pragma unroll
    for (int j = 0; j < 8; ++j) {
        float inv = 1.0f / lacc[j];
        int q = qrow0 + j + rHi;
#pragma unroll
        for (int t = 0; t < 4; ++t)
            out[((size_t)(b * S + q)) * D + h * HD + t * 16 + l16] = o[t][j] * inv;
    }
}

// =====================================================================
extern "C" void kernel_launch(void* const* d_in, const int* in_sizes, int n_in,
                              void* d_out, int out_size, void* d_ws, size_t ws_size,
                              hipStream_t stream) {
    constexpr int B = 2, S = 2048, D = 1024;
    constexpr int M = B * S;  // 4096

    const float* x      = (const float*)d_in[0];
    const float* qkv_w  = (const float*)d_in[1];
    const float* qkv_b  = (const float*)d_in[2];
    const float* proj_w = (const float*)d_in[3];
    const float* proj_b = (const float*)d_in[4];
    float* out = (float*)d_out;

    // workspace: qkv bf16 [M][3D] (24 MB) | attn f32 [M][D] (16 MB)
    unsigned short* qkv = (unsigned short*)d_ws;
    float* attn = (float*)((char*)d_ws + (size_t)M * 3 * D * sizeof(unsigned short));

    // 1) fused QKV projection -> bf16
    dim3 g1(3 * D / 256, M / 64);
    gemm_bias_kernel<true><<<g1, dim3(256), 0, stream>>>(x, qkv_w, qkv_b, qkv,
                                                         M, 3 * D, D);
    // 2) causal flash attention (per b, h, 64-query block)
    dim3 g2(S / 64, 16, B);
    attention_kernel<<<g2, dim3(128), 0, stream>>>(qkv, attn);

    // 3) output projection -> f32
    dim3 g3(D / 256, M / 64);
    gemm_bias_kernel<false><<<g3, dim3(256), 0, stream>>>(attn, proj_w, proj_b,
                                                          out, M, D, D);
}
